// CrossAttentionHook_35502199669479
// MI455X (gfx1250) — compile-verified, matrix operand order checked by reading, not run on hardware
//
#include <hip/hip_runtime.h>
#include <cstdint>
#include <cstddef>

// ---------------------------------------------------------------------------
// CrossAttentionHook for MI455X (gfx1250, wave32, WMMA bf16 + async-to-LDS)
// B=4, LQ=LKV=2048, E=2048, AUG=1024, H=16, DH=128
// ---------------------------------------------------------------------------

#define B_   4
#define LQ_  2048
#define LKV_ 2048
#define E_   2048
#define AUG_ 1024
#define H_   16
#define DH_  128

typedef __attribute__((ext_vector_type(16))) __bf16 v16bf;
typedef __attribute__((ext_vector_type(8)))  float  v8f;
typedef unsigned short bfraw;  // bf16 at the host ABI level

// ---------------------------------------------------------------------------
// CDNA5 async global->LDS copy (ASYNCcnt-tracked DMA, GVS addressing:
//   lds[ldsOff] <- mem[sbase + voff]) and its wait.
// ---------------------------------------------------------------------------
static __device__ __forceinline__
void async_ld_b128(unsigned ldsOff, unsigned voff, unsigned long long sbase)
{
    asm volatile("global_load_async_to_lds_b128 %0, %1, %2"
                 :: "v"(ldsOff), "v"(voff), "s"(sbase) : "memory");
}
static __device__ __forceinline__ void wait_async0()
{
    asm volatile("s_wait_asynccnt 0x0" ::: "memory");
}

// ---------------------------------------------------------------------------
// f32 -> bf16 conversion (vectorized)
// ---------------------------------------------------------------------------
__global__ __launch_bounds__(256)
void cvt_f32_to_bf16_kernel(const float* __restrict__ src, bfraw* __restrict__ dst_, int n4)
{
    __bf16* dst = (__bf16*)dst_;
    int i = blockIdx.x * blockDim.x + threadIdx.x;
    if (i >= n4) return;
    float4 v = ((const float4*)src)[i];
    __bf16 o[4];
    o[0] = (__bf16)v.x; o[1] = (__bf16)v.y; o[2] = (__bf16)v.z; o[3] = (__bf16)v.w;
    *(uint2*)(dst + (size_t)i * 4) = *(const uint2*)o;
}

// ---------------------------------------------------------------------------
// Tiled bf16 WMMA GEMM:  C[M,N] = A[M,K] @ W[N,K]^T + bias[N]
// Block tile 128x64, BK=32, 256 threads = 8 waves, each wave 32x32 (2x2 WMMA).
// LDS double-buffered, filled with async global->LDS DMA; compute of tile i
// overlaps the DMA of tile i+1. All dims divide the tiles exactly.
// Dynamic LDS layout (offset 0): [As0 | As1 | Bs0 | Bs1]
// ---------------------------------------------------------------------------
#define GEMM_AST    40u                      // 32 + 8 bf16 row stride (80B)
#define GEMM_ABYTES (128u * GEMM_AST * 2u)   // 10240
#define GEMM_BBYTES (64u  * GEMM_AST * 2u)   // 5120
#define GEMM_LDS    (2u * GEMM_ABYTES + 2u * GEMM_BBYTES)  // 30720

template<bool F32OUT>
__global__ __launch_bounds__(256)
void gemm_bf16_wmma(void* __restrict__ Cout,
                    const bfraw* __restrict__ A_,
                    const bfraw* __restrict__ W_,
                    const float* __restrict__ bias,
                    int K, int lda, int ldw, int ldc)
{
    constexpr int BK = 32;
    constexpr unsigned AST = GEMM_AST;
    extern __shared__ char smem[];
    const unsigned asOff[2] = {0u, GEMM_ABYTES};
    const unsigned bsOff[2] = {2u * GEMM_ABYTES, 2u * GEMM_ABYTES + GEMM_BBYTES};

    const int tid  = threadIdx.x;
    const int lane = tid & 31;
    const int wave = tid >> 5;
    const int wm = wave & 3;     // wave tile row (0..3)
    const int wn = wave >> 2;    // wave tile col (0..1)
    const int ln = lane & 15;
    const int kh = lane >> 4;

    const int rowBase = blockIdx.y * 128;
    const int colBase = blockIdx.x * 64;

    const unsigned long long Ab = (unsigned long long)A_;
    const unsigned long long Wb = (unsigned long long)W_;

    // Fixed per-thread staging coordinates (b128 = 8 bf16 per request):
    const int ar0 = tid >> 2;            // A rows 0..63
    const int ar1 = ar0 + 64;            // A rows 64..127
    const int ac  = (tid & 3) * 8;       // A/B col within BK
    const int br  = tid >> 2;            // B(W) rows 0..63

    auto stage = [&](int k0, int buf) {
        unsigned go0 = ((unsigned)(rowBase + ar0) * (unsigned)lda + (unsigned)(k0 + ac)) * 2u;
        async_ld_b128(asOff[buf] + (unsigned)(ar0 * AST + ac) * 2u, go0, Ab);
        unsigned go1 = ((unsigned)(rowBase + ar1) * (unsigned)lda + (unsigned)(k0 + ac)) * 2u;
        async_ld_b128(asOff[buf] + (unsigned)(ar1 * AST + ac) * 2u, go1, Ab);
        unsigned gob = ((unsigned)(colBase + br) * (unsigned)ldw + (unsigned)(k0 + ac)) * 2u;
        async_ld_b128(bsOff[buf] + (unsigned)(br * AST + ac) * 2u, gob, Wb);
    };

    v8f acc[2][2];
#pragma unroll
    for (int mt = 0; mt < 2; ++mt)
#pragma unroll
        for (int nt = 0; nt < 2; ++nt)
            acc[mt][nt] = (v8f){0.f,0.f,0.f,0.f,0.f,0.f,0.f,0.f};

    stage(0, 0);
    int buf = 0;
    for (int k0 = 0; k0 < K; k0 += BK) {
        wait_async0();        // this wave's DMA into buf has landed
        __syncthreads();      // everyone's DMA landed; prev compute done
        if (k0 + BK < K) stage(k0 + BK, buf ^ 1);   // overlap next DMA

        const __bf16* As = (const __bf16*)(smem + asOff[buf]);
        const __bf16* Bs = (const __bf16*)(smem + bsOff[buf]);

        // A fragments: lane holds row (ln), K = {kh*8..+7, 16+kh*8..+7}
        v16bf afrag[2], bfrag[2];
#pragma unroll
        for (int mt = 0; mt < 2; ++mt) {
            const __bf16* s = &As[(wm * 32 + mt * 16 + ln) * AST + kh * 8];
            ((uint4*)&afrag[mt])[0] = *(const uint4*)(s);
            ((uint4*)&afrag[mt])[1] = *(const uint4*)(s + 16);
        }
#pragma unroll
        for (int nt = 0; nt < 2; ++nt) {
            const __bf16* s = &Bs[(wn * 32 + nt * 16 + ln) * AST + kh * 8];
            ((uint4*)&bfrag[nt])[0] = *(const uint4*)(s);
            ((uint4*)&bfrag[nt])[1] = *(const uint4*)(s + 16);
        }
#pragma unroll
        for (int mt = 0; mt < 2; ++mt)
#pragma unroll
            for (int nt = 0; nt < 2; ++nt)
                acc[mt][nt] = __builtin_amdgcn_wmma_f32_16x16x32_bf16(
                    false, afrag[mt], false, bfrag[nt],
                    (short)0, acc[mt][nt], false, false);
        buf ^= 1;
    }

    // D layout: lane n = ln, rows m = kh*8 + r
#pragma unroll
    for (int mt = 0; mt < 2; ++mt) {
#pragma unroll
        for (int nt = 0; nt < 2; ++nt) {
            const int col  = colBase + wn * 32 + nt * 16 + ln;
            const float bv = bias[col];
            const int row0 = rowBase + wm * 32 + mt * 16 + kh * 8;
#pragma unroll
            for (int r = 0; r < 8; ++r) {
                float v = acc[mt][nt][r] + bv;
                if constexpr (F32OUT)
                    ((float*)Cout)[(size_t)(row0 + r) * ldc + col] = v;
                else
                    ((__bf16*)Cout)[(size_t)(row0 + r) * ldc + col] = (__bf16)v;
            }
        }
    }
}

// ---------------------------------------------------------------------------
// Flash attention: one block = (64 q rows, one head, one batch), 4 waves.
// Each wave owns 16 q rows. kv in 64-wide tiles with online softmax.
// K tile staged with async DMA (overlapped with the V load+transpose).
// Dynamic LDS layout: [Kt 17408B | Vt 18432B | Pw 9216B] = 45056B
// ---------------------------------------------------------------------------
#define FLA_KST   136        // DH + 8 bf16 row stride (272B)
#define FLA_VST   72         // 64 + 8 bf16 row stride (144B)
#define FLA_KT_B  (64u * FLA_KST * 2u)     // 17408
#define FLA_VT_B  (128u * FLA_VST * 2u)    // 18432
#define FLA_PW_B  (4u * 16u * FLA_VST * 2u)// 9216
#define FLA_LDS   (FLA_KT_B + FLA_VT_B + FLA_PW_B)

__global__ __launch_bounds__(128)
void flash_attn_wmma(bfraw* __restrict__ ctx_,
                     const bfraw* __restrict__ Q_,
                     const bfraw* __restrict__ K_,
                     const bfraw* __restrict__ V_)
{
    constexpr int TK  = 64;
    constexpr int KST = FLA_KST;
    constexpr int VST = FLA_VST;
    extern __shared__ char smem[];
    __bf16* Kt = (__bf16*)(smem);
    __bf16* Vt = (__bf16*)(smem + FLA_KT_B);
    __bf16* Pw = (__bf16*)(smem + FLA_KT_B + FLA_VT_B);

    __bf16* ctx = (__bf16*)ctx_;
    const __bf16* Q  = (const __bf16*)Q_;
    const __bf16* Vd = (const __bf16*)V_;
    const unsigned long long Kb = (unsigned long long)K_;

    const int tid  = threadIdx.x;
    const int lane = tid & 31;
    const int wave = tid >> 5;
    const int ln = lane & 15;
    const int kh = lane >> 4;

    const int q0 = blockIdx.x * 64;
    const int h  = blockIdx.y;
    const int b  = blockIdx.z;
    const float scale = 0.08838834764831845f;   // 1/sqrt(128)

    // Preload Q fragments for this wave's 16 rows (DH=128 -> 4 K-steps of 32)
    v16bf qfrag[4];
    {
        const __bf16* qrow = Q + ((size_t)b * LQ_ + q0 + wave * 16 + ln) * E_ + h * DH_;
#pragma unroll
        for (int ks = 0; ks < 4; ++ks) {
            const __bf16* s = qrow + ks * 32 + kh * 8;
            ((uint4*)&qfrag[ks])[0] = *(const uint4*)(s);
            ((uint4*)&qfrag[ks])[1] = *(const uint4*)(s + 16);
        }
    }

    v8f ctxacc[8];
#pragma unroll
    for (int i = 0; i < 8; ++i) ctxacc[i] = (v8f){0.f,0.f,0.f,0.f,0.f,0.f,0.f,0.f};
    float mrow[8], lrow[8];
#pragma unroll
    for (int r = 0; r < 8; ++r) { mrow[r] = -1e30f; lrow[r] = 0.0f; }

    __bf16* PwW = &Pw[wave * 16 * VST];
    const unsigned pwBase = FLA_KT_B + FLA_VT_B + (unsigned)(wave * 16 * VST) * 2u;
    (void)pwBase;

    for (int kv0 = 0; kv0 < LKV_; kv0 += TK) {
        __syncthreads();   // previous iteration's compute done before overwrite

        // 1) Kick off K-tile DMA (64x128 bf16 = 1024 b128, 8 per thread).
#pragma unroll
        for (int j = 0; j < 8; ++j) {
            int i = tid + j * 128;
            int r = i >> 4, c = (i & 15) * 8;
            unsigned go = ((unsigned)(b * LKV_ + kv0 + r) * (unsigned)E_ +
                           (unsigned)(h * DH_ + c)) * 2u;
            async_ld_b128((unsigned)(r * KST + c) * 2u, go, Kb);
        }
        // 2) Meanwhile load+transpose the V tile through VGPRs.
#pragma unroll
        for (int j = 0; j < 8; ++j) {
            int i = tid + j * 128;
            int r = i >> 4, c = (i & 15) * 8;
            const size_t gro = ((size_t)b * LKV_ + kv0 + r) * E_ + h * DH_ + c;
            uint4 vv = *(const uint4*)(Vd + gro);
            __bf16 vt[8];
            *(uint4*)vt = vv;
#pragma unroll
            for (int jj = 0; jj < 8; ++jj)
                Vt[(c + jj) * VST + r] = vt[jj];
        }
        wait_async0();
        __syncthreads();

        // S = Q @ K^T : 4 tiles of 16x16, K-dim = DH (4 WMMA steps each)
        v8f sacc[4];
#pragma unroll
        for (int nt = 0; nt < 4; ++nt) {
            v8f a = (v8f){0.f,0.f,0.f,0.f,0.f,0.f,0.f,0.f};
#pragma unroll
            for (int ks = 0; ks < 4; ++ks) {
                v16bf bf;
                const __bf16* s = &Kt[(nt * 16 + ln) * KST + ks * 32 + kh * 8];
                ((uint4*)&bf)[0] = *(const uint4*)(s);
                ((uint4*)&bf)[1] = *(const uint4*)(s + 16);
                a = __builtin_amdgcn_wmma_f32_16x16x32_bf16(
                        false, qfrag[ks], false, bf, (short)0, a, false, false);
            }
            sacc[nt] = a;
        }

        // Online softmax. Row m = kh*8 + r lives across the 16 lanes sharing kh.
#pragma unroll
        for (int r = 0; r < 8; ++r) {
            float m = sacc[0][r];
#pragma unroll
            for (int nt = 1; nt < 4; ++nt) m = fmaxf(m, sacc[nt][r]);
            m *= scale;
#pragma unroll
            for (int off = 8; off >= 1; off >>= 1)
                m = fmaxf(m, __shfl_xor(m, off, 16));
            const float mnew  = fmaxf(mrow[r], m);
            const float alpha = __expf(mrow[r] - mnew);
            mrow[r] = mnew;
            float ssum = 0.f;
#pragma unroll
            for (int nt = 0; nt < 4; ++nt) {
                float pv = __expf(sacc[nt][r] * scale - mnew);
                sacc[nt][r] = pv;
                ssum += pv;
            }
#pragma unroll
            for (int off = 8; off >= 1; off >>= 1)
                ssum += __shfl_xor(ssum, off, 16);
            lrow[r] = lrow[r] * alpha + ssum;
#pragma unroll
            for (int dt = 0; dt < 8; ++dt) ctxacc[dt][r] *= alpha;
        }

        // Re-layout P (C/D layout -> A layout) through per-wave LDS.
#pragma unroll
        for (int nt = 0; nt < 4; ++nt)
#pragma unroll
            for (int r = 0; r < 8; ++r)
                PwW[(kh * 8 + r) * VST + nt * 16 + ln] = (__bf16)sacc[nt][r];

        asm volatile("s_wait_dscnt 0x0" ::: "memory");   // within-wave DS RAW fence

        v16bf pfrag[2];
#pragma unroll
        for (int ks = 0; ks < 2; ++ks) {
            const __bf16* s = &PwW[ln * VST + ks * 32 + kh * 8];
            ((uint4*)&pfrag[ks])[0] = *(const uint4*)(s);
            ((uint4*)&pfrag[ks])[1] = *(const uint4*)(s + 16);
        }

        // ctx += P @ V : 8 d-tiles x 2 K-steps (B frags from transposed Vt)
#pragma unroll
        for (int dt = 0; dt < 8; ++dt) {
#pragma unroll
            for (int ks = 0; ks < 2; ++ks) {
                v16bf vf;
                const __bf16* s = &Vt[(dt * 16 + ln) * VST + ks * 32 + kh * 8];
                ((uint4*)&vf)[0] = *(const uint4*)(s);
                ((uint4*)&vf)[1] = *(const uint4*)(s + 16);
                ctxacc[dt] = __builtin_amdgcn_wmma_f32_16x16x32_bf16(
                        false, pfrag[ks], false, vf, (short)0, ctxacc[dt], false, false);
            }
        }
    }

    // Normalize and store ctx (bf16, [B, LQ, E] with head offset h*DH)
    __bf16* crow = ctx + ((size_t)b * LQ_ + q0 + wave * 16) * E_ + h * DH_;
#pragma unroll
    for (int r = 0; r < 8; ++r) {
        const float inv = 1.0f / lrow[r];
        const int m = kh * 8 + r;
#pragma unroll
        for (int dt = 0; dt < 8; ++dt)
            crow[(size_t)m * E_ + dt * 16 + ln] = (__bf16)(ctxacc[dt][r] * inv);
    }
}

// ---------------------------------------------------------------------------
// GemmaRMSNorm (fp32, (1+w) gain) + residual add. One block per row of E=2048.
// ---------------------------------------------------------------------------
__global__ __launch_bounds__(256)
void rmsnorm_residual_kernel(const float* __restrict__ x,
                             const float* __restrict__ res,
                             const float* __restrict__ w,
                             float* __restrict__ out)
{
    __shared__ float red[8];
    const int row = blockIdx.x;
    const int tid = threadIdx.x;
    const float4* xr = (const float4*)(x   + (size_t)row * E_);
    const float4* rr = (const float4*)(res + (size_t)row * E_);
    const float4* wv = (const float4*)w;
    float4* orow     = (float4*)(out + (size_t)row * E_);

    float ss = 0.f;
#pragma unroll
    for (int j = 0; j < (E_ / 4) / 256; ++j) {
        float4 v = xr[tid + j * 256];
        ss += v.x * v.x + v.y * v.y + v.z * v.z + v.w * v.w;
    }
#pragma unroll
    for (int off = 16; off >= 1; off >>= 1) ss += __shfl_xor(ss, off, 32);
    if ((tid & 31) == 0) red[tid >> 5] = ss;
    __syncthreads();
    if (tid == 0) {
        float t = 0.f;
#pragma unroll
        for (int i = 0; i < 8; ++i) t += red[i];
        red[0] = t;
    }
    __syncthreads();
    const float inv = rsqrtf(red[0] * (1.0f / (float)E_) + 1e-6f);

#pragma unroll
    for (int j = 0; j < (E_ / 4) / 256; ++j) {
        int i = tid + j * 256;
        float4 v = xr[i], q = rr[i], g = wv[i];
        float4 o;
        o.x = v.x * inv * (1.0f + g.x) + q.x;
        o.y = v.y * inv * (1.0f + g.y) + q.y;
        o.z = v.z * inv * (1.0f + g.z) + q.z;
        o.w = v.w * inv * (1.0f + g.w) + q.w;
        orow[i] = o;
    }
}

// ---------------------------------------------------------------------------
// Host driver
// ---------------------------------------------------------------------------
extern "C" void kernel_launch(void* const* d_in, const int* in_sizes, int n_in,
                              void* d_out, int out_size, void* d_ws, size_t ws_size,
                              hipStream_t stream)
{
    const float* query  = (const float*)d_in[0];
    const float* aug    = (const float*)d_in[1];
    // d_in[2] = aug_mask (unused by reference)
    const float* proj_w = (const float*)d_in[3];
    const float* proj_b = (const float*)d_in[4];
    const float* ipw    = (const float*)d_in[5];
    const float* ipb    = (const float*)d_in[6];
    const float* opw    = (const float*)d_in[7];
    const float* opb    = (const float*)d_in[8];
    const float* rmsw   = (const float*)d_in[9];

    const size_t MQ = (size_t)B_ * LQ_;   // 8192 rows (batch folded into M)

    char* p = (char*)d_ws;
    auto carve_bf = [&](size_t elems) -> bfraw* {
        bfraw* r = (bfraw*)p;
        p += ((elems * 2 + 255) & ~(size_t)255);
        return r;
    };
    bfraw* qryb = carve_bf(MQ * E_);
    bfraw* augb = carve_bf(MQ * AUG_);
    bfraw* pwb  = carve_bf((size_t)E_ * AUG_);
    bfraw* ipwb = carve_bf((size_t)3 * E_ * E_);
    bfraw* opwb = carve_bf((size_t)E_ * E_);
    bfraw* kvb  = carve_bf(MQ * E_);
    bfraw* qb   = carve_bf(MQ * E_);
    bfraw* kb   = carve_bf(MQ * E_);
    bfraw* vb   = carve_bf(MQ * E_);
    bfraw* ctxb = carve_bf(MQ * E_);
    float* attnf = (float*)p;             // MQ * E_ floats

    auto cvt = [&](const float* s, bfraw* d, size_t n) {
        int n4 = (int)(n / 4);
        cvt_f32_to_bf16_kernel<<<(n4 + 255) / 256, 256, 0, stream>>>(s, d, n4);
    };
    cvt(query,  qryb, MQ * E_);
    cvt(aug,    augb, MQ * AUG_);
    cvt(proj_w, pwb,  (size_t)E_ * AUG_);
    cvt(ipw,    ipwb, (size_t)3 * E_ * E_);
    cvt(opw,    opwb, (size_t)E_ * E_);

    const dim3 gblk(256);
    const dim3 ggrid(E_ / 64, (unsigned)(MQ / 128));   // 32 x 64 blocks

    // kv = aug @ proj_w^T + proj_b
    gemm_bf16_wmma<false><<<ggrid, gblk, GEMM_LDS, stream>>>(
        (void*)kvb, augb, pwb, proj_b, AUG_, AUG_, AUG_, E_);
    // q / k / v projections
    gemm_bf16_wmma<false><<<ggrid, gblk, GEMM_LDS, stream>>>(
        (void*)qb, qryb, ipwb, ipb, E_, E_, E_, E_);
    gemm_bf16_wmma<false><<<ggrid, gblk, GEMM_LDS, stream>>>(
        (void*)kb, kvb, ipwb + (size_t)E_ * E_, ipb + E_, E_, E_, E_, E_);
    gemm_bf16_wmma<false><<<ggrid, gblk, GEMM_LDS, stream>>>(
        (void*)vb, kvb, ipwb + (size_t)2 * E_ * E_, ipb + 2 * E_, E_, E_, E_, E_);

    // flash attention
    flash_attn_wmma<<<dim3(LQ_ / 64, H_, B_), dim3(128), FLA_LDS, stream>>>(
        ctxb, qb, kb, vb);

    // out projection (fp32 output for the norm)
    gemm_bf16_wmma<true><<<ggrid, gblk, GEMM_LDS, stream>>>(
        (void*)attnf, ctxb, opwb, opb, E_, E_, E_, E_);

    // RMS norm + residual into d_out
    rmsnorm_residual_kernel<<<(unsigned)MQ, 256, 0, stream>>>(
        attnf, query, rmsw, (float*)d_out);
}
